// INTERNLM_10101763080809
// MI455X (gfx1250) — matrix operand order repaired and checked by reading, not compiled, not run on hardware
//
#include <hip/hip_runtime.h>
#include <hip/hip_bf16.h>

// ---------------------------------------------------------------------------
// Model constants (fixed by the reference setup)
// ---------------------------------------------------------------------------
#define LAYERS 2
#define NH     32
#define KVH    8
#define GQA    4        // NH / KVH
#define HD     128      // head dim D
#define DM     4096
#define FF     8192
#define VOCAB  32000
#define TOK    256      // T (== S, history_len = 0)
#define EPSV   1e-6f

typedef __attribute__((ext_vector_type(16))) __bf16 bf16x16;
typedef __attribute__((ext_vector_type(8)))  float  f32x8;

// --- CDNA5 async global->LDS staging (guarded: falls back to VGPR staging) ---
#if defined(__HIP_DEVICE_COMPILE__) && defined(__has_builtin)
#  if __has_builtin(__builtin_amdgcn_global_load_async_to_lds_b128)
#    define USE_ASYNC 1
#  endif
#endif
#ifndef USE_ASYNC
#  define USE_ASYNC 0
#endif

#if USE_ASYNC
#define AS1 __attribute__((address_space(1)))
#define AS3 __attribute__((address_space(3)))
typedef int v4i_ __attribute__((vector_size(16)));   // matches builtin's pointee type
__device__ __forceinline__ void async_b128(const float* g, float* l) {
    __builtin_amdgcn_global_load_async_to_lds_b128(
        (AS1 v4i_*)(AS1 void*)g, (AS3 v4i_*)(AS3 void*)l, 0, 0);
}
template<int N>
__device__ __forceinline__ void wait_asyncN() {
#if __has_builtin(__builtin_amdgcn_s_wait_asynccnt)
    __builtin_amdgcn_s_wait_asynccnt(N);
#else
    asm volatile("s_wait_asynccnt %0" :: "i"(N) : "memory");
#endif
}
#endif

// ---------------------------------------------------------------------------
// Generic batched GEMM:  Y[z] (M x N) = A[z] (M x K) * B[z] (K x N)  [+= if ADD]
//   BT=true : B is stored [N][K] row-major (weight W, compute X @ W^T)
//   BT=false: B is stored [K][N] row-major
// Requirements: M % 128 == 0, N % 128 == 0, K % 32 == 0 (all call sites satisfy).
// 256-thread block (8 waves) computes a 128x128 tile; each wave owns a 32x64
// strip = 2 A-frags x 4 B-frags = 8 v_wmma_f32_16x16x32_bf16 per K-step.
// fp32 tiles stream HBM -> LDS via GLOBAL_LOAD_ASYNC_TO_LDS_B128 (ASYNCcnt),
// double-buffered: tile k+1 copies overlap tile k WMMA (partial asynccnt wait,
// async loads complete in order per ISA 4.1). fp32->bf16 at fragment build.
// ---------------------------------------------------------------------------
#define BM 128
#define BN 128
#define BK 32
// fallback (bf16-in-LDS) strides
#define ALD 34
#define BLD 34
// async (f32-in-LDS) strides: rows 16B-aligned, bank-conflict-free
#define ALDF 36    // [row][k] stride in floats (36*4B per row)
#define BKN  132   // [k][n] stride in floats for !BT layout

#if USE_ASYNC
// Issue one 128x32 A tile + one B tile as 8 async b128 ops per wave.
template<bool BT>
__device__ __forceinline__ void stage_async(const float* A, const float* B,
                                            int lda, int ldb, int m0, int n0,
                                            int k0, int tid, float* af, float* bf)
{
    #pragma unroll
    for (int it = 0; it < 4; ++it) {
        int chunk = tid + it * 256;          // 0..1023
        int r  = chunk >> 3;                 // 0..127
        int c4 = (chunk & 7) * 4;            // 0,4,..,28
        async_b128(A + (long long)(m0 + r) * lda + (k0 + c4), &af[r * ALDF + c4]);
    }
    if (BT) {   // B[N][K] -> LDS [n][k]
        #pragma unroll
        for (int it = 0; it < 4; ++it) {
            int chunk = tid + it * 256;
            int r  = chunk >> 3;
            int c4 = (chunk & 7) * 4;
            async_b128(B + (long long)(n0 + r) * ldb + (k0 + c4), &bf[r * ALDF + c4]);
        }
    } else {    // B[K][N] -> LDS [k][n] (no transpose; frag gather strides)
        #pragma unroll
        for (int it = 0; it < 4; ++it) {
            int chunk = tid + it * 256;
            int k  = chunk >> 5;             // 0..31
            int n4 = (chunk & 31) * 4;       // 0,4,..,124
            async_b128(B + (long long)(k0 + k) * ldb + (n0 + n4), &bf[k * BKN + n4]);
        }
    }
}
#endif

template<bool BT, bool ADD>
__global__ __launch_bounds__(256)
void gemm_bf16_wmma(const float* __restrict__ A, int lda, long long sAz,
                    const float* __restrict__ B, int ldb, long long sBz, int zdivB,
                    float* __restrict__ Y, int ldy, long long sYz,
                    int M, int N, int K)
{
    const int z = blockIdx.z;
    A += (long long)z * sAz;
    B += (long long)(z / zdivB) * sBz;
    Y += (long long)z * sYz;

    const int m0   = blockIdx.y * BM;
    const int n0   = blockIdx.x * BN;
    const int tid  = threadIdx.x;
    const int lane = tid & 31;
    const int wave = tid >> 5;
    const int wm   = wave >> 1;      // 0..3 : 32-row strip within tile
    const int wn   = wave & 1;       // 0..1 : 64-col strip within tile
    const int lLo  = lane & 15;
    const int hiS  = lane >> 4;      // lane half select

    f32x8 acc[2][4] = {};

#if USE_ASYNC
    __shared__ __align__(16) float Af[2][BM * ALDF];       // 2 x 18 KB, [m][k]
    __shared__ __align__(16) float Bf[2][BM * ALDF];       // 2 x 18 KB

    stage_async<BT>(A, B, lda, ldb, m0, n0, 0, tid, Af[0], Bf[0]);
    int p = 0;
    for (int k0 = 0; k0 < K; k0 += BK) {
        if (k0 + BK < K) {
            // prefetch next tile into alternate buffer, then retire only the
            // current tile's 8 async ops (in-order completion)
            stage_async<BT>(A, B, lda, ldb, m0, n0, k0 + BK, tid, Af[p ^ 1], Bf[p ^ 1]);
            wait_asyncN<8>();
        } else {
            wait_asyncN<0>();
        }
        __syncthreads();

        const float* af = Af[p];
        const float* bf = Bf[p];

        // ---- fragments (k = 16*hiS + element in both A and B), f32 -> bf16 at build ----
        bf16x16 a[2];
        #pragma unroll
        for (int s = 0; s < 2; ++s) {
            const float* ap = &af[(wm * 32 + s * 16 + lLo) * ALDF];
            #pragma unroll
            for (int j = 0; j < 8; ++j) {
                int kb = 2 * j + hiS * 16;
                a[s][2 * j]     = (__bf16)ap[kb];
                a[s][2 * j + 1] = (__bf16)ap[kb + 1];
            }
        }
        bf16x16 b[4];
        #pragma unroll
        for (int s = 0; s < 4; ++s) {
            int ncol = wn * 64 + s * 16 + lLo;
            if (BT) {
                const float* bp = &bf[ncol * ALDF + hiS * 16];
                #pragma unroll
                for (int h = 0; h < 16; ++h) b[s][h] = (__bf16)bp[h];
            } else {
                #pragma unroll
                for (int h = 0; h < 16; ++h)
                    b[s][h] = (__bf16)bf[(h + hiS * 16) * BKN + ncol];
            }
        }

        #pragma unroll
        for (int i = 0; i < 2; ++i)
            #pragma unroll
            for (int j = 0; j < 4; ++j)
                acc[i][j] = __builtin_amdgcn_wmma_f32_16x16x32_bf16(
                    false, a[i], false, b[j], (short)0, acc[i][j], false, false);
        __syncthreads();
        p ^= 1;
    }
#else   // -------- fallback: VGPR staging with bf16 LDS tiles --------
    __shared__ __bf16 As[BM * ALD];
    __shared__ __bf16 Bs[BN * BLD];
    const int sr = tid >> 1;
    const int sc = (tid & 1) * 16;

    for (int k0 = 0; k0 < K; k0 += BK) {
        {
            const float* src = A + (long long)(m0 + sr) * lda + (k0 + sc);
            float4 f0 = *(const float4*)(src + 0);
            float4 f1 = *(const float4*)(src + 4);
            float4 f2 = *(const float4*)(src + 8);
            float4 f3 = *(const float4*)(src + 12);
            __bf16* dst = &As[sr * ALD + sc];
            dst[0]=(__bf16)f0.x; dst[1]=(__bf16)f0.y; dst[2]=(__bf16)f0.z; dst[3]=(__bf16)f0.w;
            dst[4]=(__bf16)f1.x; dst[5]=(__bf16)f1.y; dst[6]=(__bf16)f1.z; dst[7]=(__bf16)f1.w;
            dst[8]=(__bf16)f2.x; dst[9]=(__bf16)f2.y; dst[10]=(__bf16)f2.z; dst[11]=(__bf16)f2.w;
            dst[12]=(__bf16)f3.x; dst[13]=(__bf16)f3.y; dst[14]=(__bf16)f3.z; dst[15]=(__bf16)f3.w;
        }
        if (BT) {
            const float* src = B + (long long)(n0 + sr) * ldb + (k0 + sc);
            float4 f0 = *(const float4*)(src + 0);
            float4 f1 = *(const float4*)(src + 4);
            float4 f2 = *(const float4*)(src + 8);
            float4 f3 = *(const float4*)(src + 12);
            __bf16* dst = &Bs[sr * BLD + sc];
            dst[0]=(__bf16)f0.x; dst[1]=(__bf16)f0.y; dst[2]=(__bf16)f0.z; dst[3]=(__bf16)f0.w;
            dst[4]=(__bf16)f1.x; dst[5]=(__bf16)f1.y; dst[6]=(__bf16)f1.z; dst[7]=(__bf16)f1.w;
            dst[8]=(__bf16)f2.x; dst[9]=(__bf16)f2.y; dst[10]=(__bf16)f2.z; dst[11]=(__bf16)f2.w;
            dst[12]=(__bf16)f3.x; dst[13]=(__bf16)f3.y; dst[14]=(__bf16)f3.z; dst[15]=(__bf16)f3.w;
        } else {
            #pragma unroll
            for (int it = 0; it < (BN * BK) / 256; ++it) {
                int i = tid + it * 256;
                int c = i >> 7;
                int n = i & (BN - 1);
                Bs[n * BLD + c] = (__bf16)B[(long long)(k0 + c) * ldb + (n0 + n)];
            }
        }
        __syncthreads();

        bf16x16 a[2];
        #pragma unroll
        for (int s = 0; s < 2; ++s) {
            const __bf16* ap = &As[(wm * 32 + s * 16 + lLo) * ALD];
            #pragma unroll
            for (int j = 0; j < 8; ++j) {
                int kb = 2 * j + hiS * 16;
                a[s][2 * j]     = ap[kb];
                a[s][2 * j + 1] = ap[kb + 1];
            }
        }
        bf16x16 b[4];
        #pragma unroll
        for (int s = 0; s < 4; ++s) {
            const __bf16* bp = &Bs[(wn * 64 + s * 16 + lLo) * BLD + hiS * 16];
            #pragma unroll
            for (int h = 0; h < 16; ++h) b[s][h] = bp[h];
        }

        #pragma unroll
        for (int i = 0; i < 2; ++i)
            #pragma unroll
            for (int j = 0; j < 4; ++j)
                acc[i][j] = __builtin_amdgcn_wmma_f32_16x16x32_bf16(
                    false, a[i], false, b[j], (short)0, acc[i][j], false, false);
        __syncthreads();
    }
#endif

    // D/C layout: vgpr r -> M = +r + 8*hiSel, N = lLo
    #pragma unroll
    for (int i = 0; i < 2; ++i) {
        #pragma unroll
        for (int j = 0; j < 4; ++j) {
            #pragma unroll
            for (int r = 0; r < 8; ++r) {
                int m = m0 + wm * 32 + i * 16 + r + hiS * 8;
                int n = n0 + wn * 64 + j * 16 + lLo;
                long long idx = (long long)m * ldy + n;
                if (ADD) Y[idx] += acc[i][j][r];
                else     Y[idx]  = acc[i][j][r];
            }
        }
    }
}

// ---------------------------------------------------------------------------
// Elementwise / reduction kernels
// ---------------------------------------------------------------------------
__global__ __launch_bounds__(256)
void embed_kernel(const int* __restrict__ ids, const int* __restrict__ eq,
                  const float* __restrict__ scale, const float* __restrict__ zp,
                  float* __restrict__ hidden)
{
    int idx = blockIdx.x * 256 + threadIdx.x;     // t*DM + d
    int t = idx / DM, d = idx - t * DM;
    int id = ids[t];
    hidden[idx] = (float)eq[(long long)id * DM + d] * scale[id] + zp[id];
}

__global__ __launch_bounds__(256)
void rmsnorm_kernel(const float* __restrict__ X, const float* __restrict__ w,
                    float* __restrict__ Y)
{
    __shared__ float red[256];
    const int r = blockIdx.x;
    const float* x = X + (long long)r * DM;
    float*       y = Y + (long long)r * DM;
    float ss = 0.f;
    for (int d = threadIdx.x; d < DM; d += 256) { float v = x[d]; ss += v * v; }
    red[threadIdx.x] = ss;
    __syncthreads();
    for (int s = 128; s > 0; s >>= 1) {
        if (threadIdx.x < s) red[threadIdx.x] += red[threadIdx.x + s];
        __syncthreads();
    }
    float inv = rsqrtf(red[0] / (float)DM + EPSV);
    for (int d = threadIdx.x; d < DM; d += 256) y[d] = w[d] * x[d] * inv;
}

// Q RoPE: q [T][DM] -> qr [NH][T][HD]
__global__ __launch_bounds__(256)
void rope_q_kernel(const float* __restrict__ q, const float* __restrict__ cosr,
                   const float* __restrict__ sinr, float* __restrict__ qr)
{
    int idx = blockIdx.x * 256 + threadIdx.x;     // t*NH*HD + h*HD + d
    int d = idx & (HD - 1);
    int h = (idx / HD) & (NH - 1);
    int t = idx / (HD * NH);
    const float* row = q + (long long)t * DM + h * HD;
    float x  = row[d];
    float xo = (d < HD / 2) ? -row[d + HD / 2] : row[d - HD / 2];
    float v  = x * cosr[t * HD + d] + xo * sinr[t * HD + d];
    qr[((long long)h * TOK + t) * HD + d] = v;
}

// K RoPE + write straight into save_k cache region: [KVH][HD][S]
__global__ __launch_bounds__(256)
void rope_k_kernel(const float* __restrict__ k, const float* __restrict__ cosr,
                   const float* __restrict__ sinr, float* __restrict__ kout)
{
    int idx = blockIdx.x * 256 + threadIdx.x;     // t*KVH*HD + kh*HD + d
    int d  = idx & (HD - 1);
    int kh = (idx / HD) & (KVH - 1);
    int t  = idx / (HD * KVH);
    const float* row = k + (long long)t * (KVH * HD) + kh * HD;
    float x  = row[d];
    float xo = (d < HD / 2) ? -row[d + HD / 2] : row[d - HD / 2];
    float v  = x * cosr[t * HD + d] + xo * sinr[t * HD + d];
    kout[((long long)kh * HD + d) * TOK + t] = v;
}

// V copy into save_v cache region: [KVH][S][HD]
__global__ __launch_bounds__(256)
void copy_v_kernel(const float* __restrict__ v, float* __restrict__ vout)
{
    int idx = blockIdx.x * 256 + threadIdx.x;     // t*KVH*HD + kh*HD + d
    int d  = idx & (HD - 1);
    int kh = (idx / HD) & (KVH - 1);
    int t  = idx / (HD * KVH);
    vout[((long long)kh * TOK + t) * HD + d] = v[(long long)t * (KVH * HD) + kh * HD + d];
}

// Causal-masked softmax over rows of scores [NH*T][S], in place. S == 256.
__global__ __launch_bounds__(256)
void softmax_kernel(float* __restrict__ s, const int* __restrict__ mask_flag)
{
    __shared__ float red[256];
    const int row = blockIdx.x;
    const int t   = row & (TOK - 1);
    const int c   = threadIdx.x;
    float mf = (float)(*mask_flag);
    float v  = s[(long long)row * TOK + c] + ((c > t) ? -128.0f * mf : 0.0f);

    red[c] = v;
    __syncthreads();
    for (int off = 128; off > 0; off >>= 1) {
        if (c < off) red[c] = fmaxf(red[c], red[c + off]);
        __syncthreads();
    }
    float mx = red[0];
    __syncthreads();
    float e = __expf(v - mx);
    red[c] = e;
    __syncthreads();
    for (int off = 128; off > 0; off >>= 1) {
        if (c < off) red[c] += red[c + off];
        __syncthreads();
    }
    s[(long long)row * TOK + c] = e / red[0];
}

// g = silu(g) * u  (in place on g)
__global__ __launch_bounds__(256)
void silu_mul_kernel(float* __restrict__ g, const float* __restrict__ u)
{
    int idx = blockIdx.x * 256 + threadIdx.x;
    float x = g[idx];
    g[idx] = (x / (1.0f + __expf(-x))) * u[idx];
}

// logits[v] = dot(x, W[v]) ; one wave per output row
__global__ __launch_bounds__(256)
void lm_head_kernel(const float* __restrict__ x, const float* __restrict__ w,
                    float* __restrict__ logits)
{
    int v    = blockIdx.x * 8 + (threadIdx.x >> 5);
    int lane = threadIdx.x & 31;
    const float* wr = w + (long long)v * DM;
    float s = 0.f;
    for (int d = lane * 4; d < DM; d += 32 * 4) {
        float4 xv = *(const float4*)(x + d);
        float4 wv = *(const float4*)(wr + d);
        s += xv.x * wv.x + xv.y * wv.y + xv.z * wv.z + xv.w * wv.w;
    }
    #pragma unroll
    for (int off = 16; off > 0; off >>= 1) s += __shfl_down(s, off, 32);
    if (lane == 0) logits[v] = s;
}

__global__ __launch_bounds__(256)
void argmax_kernel(const float* __restrict__ logits, int* __restrict__ tok)
{
    __shared__ float bv[256];
    __shared__ int   bi[256];
    float best = -3.4e38f;
    int   bidx = 0;
    for (int v = threadIdx.x; v < VOCAB; v += 256) {
        float x = logits[v];
        if (x > best) { best = x; bidx = v; }
    }
    bv[threadIdx.x] = best;
    bi[threadIdx.x] = bidx;
    __syncthreads();
    for (int off = 128; off > 0; off >>= 1) {
        if (threadIdx.x < off) {
            float ov = bv[threadIdx.x + off];
            int   oi = bi[threadIdx.x + off];
            if (ov > bv[threadIdx.x] ||
                (ov == bv[threadIdx.x] && oi < bi[threadIdx.x])) {
                bv[threadIdx.x] = ov;
                bi[threadIdx.x] = oi;
            }
        }
        __syncthreads();
    }
    if (threadIdx.x == 0) *tok = bi[0];
}

// ---------------------------------------------------------------------------
// Host orchestration
// ---------------------------------------------------------------------------
extern "C" void kernel_launch(void* const* d_in, const int* in_sizes, int n_in,
                              void* d_out, int out_size, void* d_ws, size_t ws_size,
                              hipStream_t stream)
{
    (void)in_sizes; (void)n_in; (void)out_size; (void)ws_size;

    const int*   ids    = (const int*)  d_in[0];
    // d_in[1], d_in[2]: empty kv caches (history_len == 0)
    const int*   eq     = (const int*)  d_in[3];
    const float* scale  = (const float*)d_in[4];
    const float* zp     = (const float*)d_in[5];
    const float* wq     = (const float*)d_in[6];
    const float* wk     = (const float*)d_in[7];
    const float* wv     = (const float*)d_in[8];
    const float* wo     = (const float*)d_in[9];
    const float* wg     = (const float*)d_in[10];
    const float* wu     = (const float*)d_in[11];
    const float* wd     = (const float*)d_in[12];
    const float* ln1    = (const float*)d_in[13];
    const float* ln2    = (const float*)d_in[14];
    const float* normw  = (const float*)d_in[15];
    const float* lmw    = (const float*)d_in[16];
    const float* cosr   = (const float*)d_in[17];
    const float* sinr   = (const float*)d_in[18];
    const int*   maskf  = (const int*)  d_in[21];

    // Output layout: logits[32000] | next_token int32 [1] | save_k | save_v
    float* out_logits = (float*)d_out;
    int*   out_tok    = (int*)d_out + VOCAB;
    float* out_k      = (float*)d_out + VOCAB + 1;
    float* out_v      = out_k + (long long)LAYERS * KVH * HD * TOK;

    // Workspace carve-up (floats); ~46 MB total
    float* ws = (float*)d_ws;
    size_t o = 0;
    float* hidden = ws + o; o += (size_t)TOK * DM;          // 4 MB
    float* xn     = ws + o; o += (size_t)TOK * DM;          // 4 MB
    float* qbuf   = ws + o; o += (size_t)TOK * DM;          // 4 MB
    float* kbuf   = ws + o; o += (size_t)TOK * KVH * HD;    // 1 MB
    float* vbuf   = ws + o; o += (size_t)TOK * KVH * HD;    // 1 MB
    float* qr     = ws + o; o += (size_t)NH * TOK * HD;     // 4 MB
    float* scores = ws + o; o += (size_t)NH * TOK * TOK;    // 8 MB
    float* attn   = ws + o; o += (size_t)TOK * DM;          // 4 MB
    float* gate   = ws + o; o += (size_t)TOK * FF;          // 8 MB
    float* up     = ws + o; o += (size_t)TOK * FF;          // 8 MB
    float* lastn  = ws + o; o += (size_t)DM;

    const dim3 blk(256);

    // Embedding dequant gather
    embed_kernel<<<dim3(TOK * DM / 256), blk, 0, stream>>>(ids, eq, scale, zp, hidden);

    for (int i = 0; i < LAYERS; ++i) {
        const float* wq_i = wq + (size_t)i * DM * DM;
        const float* wk_i = wk + (size_t)i * KVH * HD * DM;
        const float* wv_i = wv + (size_t)i * KVH * HD * DM;
        const float* wo_i = wo + (size_t)i * DM * DM;
        const float* wg_i = wg + (size_t)i * FF * DM;
        const float* wu_i = wu + (size_t)i * FF * DM;
        const float* wd_i = wd + (size_t)i * DM * FF;
        float* kc_i = out_k + (size_t)i * KVH * HD * TOK;
        float* vc_i = out_v + (size_t)i * KVH * TOK * HD;

        // xn = rmsnorm(hidden, ln1)
        rmsnorm_kernel<<<dim3(TOK), blk, 0, stream>>>(hidden, ln1 + (size_t)i * DM, xn);

        // q/k/v projections: X @ W^T
        gemm_bf16_wmma<true, false><<<dim3(DM / BN, TOK / BM, 1), blk, 0, stream>>>(
            xn, DM, 0, wq_i, DM, 0, 1, qbuf, DM, 0, TOK, DM, DM);
        gemm_bf16_wmma<true, false><<<dim3((KVH * HD) / BN, TOK / BM, 1), blk, 0, stream>>>(
            xn, DM, 0, wk_i, DM, 0, 1, kbuf, KVH * HD, 0, TOK, KVH * HD, DM);
        gemm_bf16_wmma<true, false><<<dim3((KVH * HD) / BN, TOK / BM, 1), blk, 0, stream>>>(
            xn, DM, 0, wv_i, DM, 0, 1, vbuf, KVH * HD, 0, TOK, KVH * HD, DM);

        // RoPE; K goes transposed straight into the cache output ([KVH][HD][S])
        rope_q_kernel<<<dim3(TOK * NH * HD / 256), blk, 0, stream>>>(qbuf, cosr, sinr, qr);
        rope_k_kernel<<<dim3(TOK * KVH * HD / 256), blk, 0, stream>>>(kbuf, cosr, sinr, kc_i);
        copy_v_kernel<<<dim3(TOK * KVH * HD / 256), blk, 0, stream>>>(vbuf, vc_i);

        // scores[h] = Q_h (T x D) @ K_h (D x S); K shared across GQA groups (z/4)
        gemm_bf16_wmma<false, false><<<dim3(TOK / BN, TOK / BM, NH), blk, 0, stream>>>(
            qr, HD, (long long)TOK * HD,
            kc_i, TOK, (long long)HD * TOK, GQA,
            scores, TOK, (long long)TOK * TOK,
            TOK, TOK, HD);

        // causal mask + softmax (in place)
        softmax_kernel<<<dim3(NH * TOK), blk, 0, stream>>>(scores, maskf);

        // O_h = P_h (T x S) @ V_h (S x D); write interleaved into attn[T][DM]
        gemm_bf16_wmma<false, false><<<dim3(HD / BN, TOK / BM, NH), blk, 0, stream>>>(
            scores, TOK, (long long)TOK * TOK,
            vc_i, HD, (long long)TOK * HD, GQA,
            attn, DM, (long long)HD,
            TOK, HD, TOK);

        // hidden += attn @ wo^T
        gemm_bf16_wmma<true, true><<<dim3(DM / BN, TOK / BM, 1), blk, 0, stream>>>(
            attn, DM, 0, wo_i, DM, 0, 1, hidden, DM, 0, TOK, DM, DM);

        // MLP
        rmsnorm_kernel<<<dim3(TOK), blk, 0, stream>>>(hidden, ln2 + (size_t)i * DM, xn);
        gemm_bf16_wmma<true, false><<<dim3(FF / BN, TOK / BM, 1), blk, 0, stream>>>(
            xn, DM, 0, wg_i, DM, 0, 1, gate, FF, 0, TOK, FF, DM);
        gemm_bf16_wmma<true, false><<<dim3(FF / BN, TOK / BM, 1), blk, 0, stream>>>(
            xn, DM, 0, wu_i, DM, 0, 1, up, FF, 0, TOK, FF, DM);
        silu_mul_kernel<<<dim3(TOK * FF / 256), blk, 0, stream>>>(gate, up);
        gemm_bf16_wmma<true, true><<<dim3(DM / BN, TOK / BM, 1), blk, 0, stream>>>(
            gate, FF, 0, wd_i, FF, 0, 1, hidden, DM, 0, TOK, DM, FF);
    }

    // Final norm on last token, lm head, argmax
    rmsnorm_kernel<<<dim3(1), blk, 0, stream>>>(hidden + (size_t)(TOK - 1) * DM, normw, lastn);
    lm_head_kernel<<<dim3(VOCAB / 8), blk, 0, stream>>>(lastn, lmw, out_logits);
    argmax_kernel<<<dim3(1), blk, 0, stream>>>(out_logits, out_tok);
}
